// GAT_21440476742322
// MI455X (gfx1250) — compile-verified, matrix operand order checked by reading, not run on hardware
//
#include <hip/hip_runtime.h>

#define HEADS 8
#define NEG_SLOPE 0.2f

typedef __attribute__((ext_vector_type(16))) __bf16 v16bf;
typedef __attribute__((ext_vector_type(8)))  float  v8f;
typedef unsigned int u32x4 __attribute__((ext_vector_type(4)));

// ---------- helpers ----------
static __device__ __forceinline__ unsigned short f2bf(float f) {
    unsigned u = __float_as_uint(f);
    u += 0x7FFFu + ((u >> 16) & 1u);          // round-to-nearest-even
    return (unsigned short)(u >> 16);
}
// order-preserving float<->uint map for atomicMax on floats
static __device__ __forceinline__ unsigned fenc(float f) {
    unsigned u = __float_as_uint(f);
    return (u & 0x80000000u) ? ~u : (u | 0x80000000u);
}
static __device__ __forceinline__ float fdec(unsigned u) {
    return (u & 0x80000000u) ? __uint_as_float(u ^ 0x80000000u)
                             : __uint_as_float(~u);
}
// WMMA 16-bit fragment: lane l holds row/col (l&15), K-half (l>>4).
// Elements 0..7 -> K = kh*8 + j ; elements 8..15 -> K = 16 + kh*8 + j.
// With K-major storage that's two contiguous 16-byte loads.
static __device__ __forceinline__ v16bf load_frag(const unsigned short* p) {
    union { u32x4 q[2]; v16bf v; } u;
    u.q[0] = *(const u32x4*)(p);
    u.q[1] = *(const u32x4*)(p + 16);
    return u.v;
}

// ---------- prep kernels ----------
__global__ void k_cvt_pad(const float* __restrict__ X, unsigned short* __restrict__ Xb,
                          int K, int Kp, long long total) {
    long long i = (long long)blockIdx.x * blockDim.x + threadIdx.x;
    if (i >= total) return;
    long long n = i / Kp; int k = (int)(i % Kp);
    Xb[i] = (k < K) ? f2bf(X[n * K + k]) : (unsigned short)0;
}
// WT[n*Kp+k] = W[k*Nw+n]  (transpose weights to K-contiguous rows, pad K)
__global__ void k_wtrans(const float* __restrict__ W, unsigned short* __restrict__ WT,
                         int K, int Nw, int Kp, long long total) {
    long long i = (long long)blockIdx.x * blockDim.x + threadIdx.x;
    if (i >= total) return;
    int n = (int)(i / Kp); int k = (int)(i % Kp);
    WT[i] = (k < K) ? f2bf(W[(size_t)k * Nw + n]) : (unsigned short)0;
}
__global__ void k_fill_u32(unsigned* p, unsigned v, int n) {
    int i = blockIdx.x * blockDim.x + threadIdx.x;
    if (i < n) p[i] = v;
}
__global__ void k_fill_f32(float* p, float v, long long n) {
    long long i = (long long)blockIdx.x * blockDim.x + threadIdx.x;
    if (i < n) p[i] = v;
}

// ---------- bf16 WMMA GEMM: C[M,Nout] = A[M,Kp] * B[Kp,Nout] ----------
// A: bf16 row-major [M,Kp]; BT: bf16 [Nout,Kp] (i.e. B transposed, K-contiguous).
// One wave computes a 16 x (16*NT) tile, reusing the A fragment NT times.
template <int NT>
__global__ void k_gemm_bf16(const unsigned short* __restrict__ A,
                            const unsigned short* __restrict__ BT,
                            float* __restrict__ C,
                            int M16, int Kp, int Nout) {
    int wid    = blockIdx.x * (blockDim.x >> 5) + (threadIdx.x >> 5);
    int groups = Nout / (16 * NT);
    if (wid >= M16 * groups) return;              // wave-uniform: EXEC stays full
    int m0 = (wid / groups) * 16;
    int n0 = (wid % groups) * 16 * NT;
    int lane = threadIdx.x & 31;
    int r16  = lane & 15;
    int kh   = lane >> 4;

    v8f acc[NT];
#pragma unroll
    for (int t = 0; t < NT; ++t) acc[t] = (v8f){0.f,0.f,0.f,0.f,0.f,0.f,0.f,0.f};

    const unsigned short* arow = A + (size_t)(m0 + r16) * Kp + kh * 8;
    for (int k0 = 0; k0 < Kp; k0 += 32) {
        v16bf a = load_frag(arow + k0);
#pragma unroll
        for (int t = 0; t < NT; ++t) {
            const unsigned short* brow =
                BT + (size_t)(n0 + t * 16 + r16) * Kp + kh * 8 + k0;
            v16bf b = load_frag(brow);
            acc[t] = __builtin_amdgcn_wmma_f32_16x16x32_bf16(
                false, a, false, b, (short)0, acc[t], false, false);
        }
    }
    // C/D layout: lane (l&15)=N, l>>4 selects M-half; VGPR r -> M = r + 8*(l>>4)
#pragma unroll
    for (int t = 0; t < NT; ++t)
#pragma unroll
        for (int r = 0; r < 8; ++r)
            C[(size_t)(m0 + kh * 8 + r) * Nout + (n0 + t * 16 + r16)] = acc[t][r];
}

// ---------- attention scores: el/er [N,H] from fc output ----------
__global__ void k_attn(const float* __restrict__ g, int stride, int D,
                       const float* __restrict__ al, const float* __restrict__ ar,
                       float* __restrict__ el, float* __restrict__ er, int NH) {
    int i = blockIdx.x * blockDim.x + threadIdx.x;
    if (i >= NH) return;
    int n = i / HEADS, h = i % HEADS;
    const float* row = g + (size_t)n * stride + h * D;
    float sl = 0.f, sr = 0.f;
    for (int d = 0; d < D; ++d) {
        float v = row[d];
        sl += v * al[h * D + d];
        sr += v * ar[h * D + d];
    }
    el[i] = sl; er[i] = sr;
}

// ---------- edge pass 1: leaky-relu score + per-dst max ----------
__global__ void k_edge_max(const int* __restrict__ src, const int* __restrict__ dst,
                           const float* __restrict__ el, const float* __restrict__ er,
                           float* __restrict__ e_ws, unsigned* __restrict__ m, int EH) {
    int i = blockIdx.x * blockDim.x + threadIdx.x;
    if (i >= EH) return;
    int e = i / HEADS, h = i % HEADS;
    int s = src[e], d = dst[e];
    float v = el[s * HEADS + h] + er[d * HEADS + h];
    v = (v > 0.f) ? v : NEG_SLOPE * v;
    e_ws[i] = v;
    atomicMax(&m[d * HEADS + h], fenc(v));
}

// ---------- edge pass 2: exp(e - max) + per-dst sum ----------
__global__ void k_edge_expsum(const int* __restrict__ dst, float* __restrict__ e_ws,
                              const unsigned* __restrict__ m, float* __restrict__ ssum,
                              int EH) {
    int i = blockIdx.x * blockDim.x + threadIdx.x;
    if (i >= EH) return;
    int e = i / HEADS, h = i % HEADS;
    int d = dst[e];
    float ee = expf(e_ws[i] - fdec(m[d * HEADS + h]));
    e_ws[i] = ee;
    atomicAdd(&ssum[d * HEADS + h], ee);
}

// ---------- edge pass 3: alpha-weighted message scatter ----------
__global__ void k_edge_scatter(const int* __restrict__ src, const int* __restrict__ dst,
                               const float* __restrict__ e_ws, const float* __restrict__ ssum,
                               const float* __restrict__ g, int gStride, int D,
                               float* __restrict__ acc, long long total) {
    long long i = (long long)blockIdx.x * blockDim.x + threadIdx.x;
    if (i >= total) return;
    int HD = HEADS * D;
    int e = (int)(i / HD);
    int c = (int)(i % HD);
    int h = c / D;
    int s = src[e], d = dst[e];
    float alpha = e_ws[e * HEADS + h] / ssum[d * HEADS + h];
    atomicAdd(&acc[(size_t)d * HD + c], g[(size_t)s * gStride + c] * alpha);
}

// ---------- epilogue: out = act(acc + resid + bias) ----------
__global__ void k_epilogue(float* __restrict__ out, const float* __restrict__ acc,
                           const float* __restrict__ resid, int resStride,
                           const float* __restrict__ bias, int HD, int doElu,
                           long long total) {
    long long i = (long long)blockIdx.x * blockDim.x + threadIdx.x;
    if (i >= total) return;
    int n = (int)(i / HD), c = (int)(i % HD);
    float v = acc[i] + bias[c];
    if (resid) v += resid[(size_t)n * resStride + c];
    if (doElu) v = (v > 0.f) ? v : (expf(v) - 1.f);
    out[i] = v;
}

// ---------- mean over heads ----------
__global__ void k_mean(const float* __restrict__ h3, float* __restrict__ out,
                       int O, int total) {
    int i = blockIdx.x * blockDim.x + threadIdx.x;
    if (i >= total) return;
    int n = i / O, o = i % O;
    float s = 0.f;
    for (int h = 0; h < HEADS; ++h) s += h3[(size_t)n * HEADS * O + h * O + o];
    out[i] = s * (1.f / HEADS);
}

extern "C" void kernel_launch(void* const* d_in, const int* in_sizes, int n_in,
                              void* d_out, int out_size, void* d_ws, size_t ws_size,
                              hipStream_t stream) {
    const float* x   = (const float*)d_in[0];
    const int*   src = (const int*)d_in[1];
    const int*   dst = (const int*)d_in[2];
    const float* W1  = (const float*)d_in[3];
    const float* al1 = (const float*)d_in[4];
    const float* ar1 = (const float*)d_in[5];
    const float* b1  = (const float*)d_in[6];
    const float* W2  = (const float*)d_in[7];
    const float* al2 = (const float*)d_in[8];
    const float* ar2 = (const float*)d_in[9];
    const float* b2  = (const float*)d_in[10];
    const float* W3  = (const float*)d_in[11];
    const float* al3 = (const float*)d_in[12];
    const float* ar3 = (const float*)d_in[13];
    const float* b3  = (const float*)d_in[14];
    const float* rW3 = (const float*)d_in[15];

    const int IN = 300, KP1 = 320, HID = 16, OUT = 2;
    const int HD = HEADS * HID;   // 128
    const int HO = HEADS * OUT;   // 16
    const int N  = in_sizes[0] / IN;
    const int E  = in_sizes[1];
    const int M16 = N / 16;

    // workspace layout (all 256B aligned)
    char* base = (char*)d_ws;
    size_t off = 0;
    auto take = [&](size_t bytes) -> char* {
        char* p = base + off;
        off += (bytes + 255) & ~(size_t)255;
        return p;
    };
    unsigned short* xb  = (unsigned short*)take((size_t)N * KP1 * 2);
    unsigned short* w1t = (unsigned short*)take((size_t)HD * KP1 * 2);
    unsigned short* w2t = (unsigned short*)take((size_t)HD * HD * 2);
    unsigned short* w3t = (unsigned short*)take((size_t)(2 * HO) * HD * 2); // [W3;resW3]^T
    float* g    = (float*)take((size_t)N * HD * 4);  // fc output
    float* acc  = (float*)take((size_t)N * HD * 4);  // scatter accumulator
    float* h1   = (float*)take((size_t)N * HD * 4);
    float* h2   = (float*)take((size_t)N * HD * 4);
    unsigned short* hb = (unsigned short*)take((size_t)N * HD * 2); // bf16 layer input
    float* el   = (float*)take((size_t)N * HEADS * 4);
    float* er   = (float*)take((size_t)N * HEADS * 4);
    unsigned* mmax = (unsigned*)take((size_t)N * HEADS * 4);
    float* ssum = (float*)take((size_t)N * HEADS * 4);
    float* ews  = (float*)take((size_t)E * HEADS * 4);
    float* h3   = (float*)hb;  // safe reuse: hb's last reader (layer-3 GEMM) precedes this
    (void)ws_size; (void)n_in; (void)out_size;

    const int B = 256;
    auto blk = [](long long n, int b) { return (unsigned)((n + b - 1) / b); };
    auto gemm_blocks = [&](int groups) { return (unsigned)(((long long)M16 * groups + 3) / 4); };
    const unsigned ENC_NEG_INF = 0x007FFFFFu; // fenc(-inf)

    // ---- prep: bf16 conversions / weight transposes ----
    k_cvt_pad<<<blk((long long)N * KP1, B), B, 0, stream>>>(x, xb, IN, KP1, (long long)N * KP1);
    k_wtrans<<<blk((long long)HD * KP1, B), B, 0, stream>>>(W1, w1t, IN, HD, KP1, (long long)HD * KP1);
    k_wtrans<<<blk((long long)HD * HD, B), B, 0, stream>>>(W2, w2t, HD, HD, HD, (long long)HD * HD);
    k_wtrans<<<blk((long long)HO * HD, B), B, 0, stream>>>(W3, w3t, HD, HO, HD, (long long)HO * HD);
    k_wtrans<<<blk((long long)HO * HD, B), B, 0, stream>>>(rW3, w3t + (size_t)HO * HD, HD, HO, HD, (long long)HO * HD);

    auto edge_phase = [&](const float* gfc, int gStride, int D,
                          const float* alp, const float* arp, float* accbuf) {
        int HDl = HEADS * D;
        k_attn<<<blk((long long)N * HEADS, B), B, 0, stream>>>(gfc, gStride, D, alp, arp, el, er, N * HEADS);
        k_fill_u32<<<blk((long long)N * HEADS, B), B, 0, stream>>>(mmax, ENC_NEG_INF, N * HEADS);
        k_fill_f32<<<blk((long long)N * HEADS, B), B, 0, stream>>>(ssum, 0.f, (long long)N * HEADS);
        k_fill_f32<<<blk((long long)N * HDl, B), B, 0, stream>>>(accbuf, 0.f, (long long)N * HDl);
        k_edge_max<<<blk((long long)E * HEADS, B), B, 0, stream>>>(src, dst, el, er, ews, mmax, E * HEADS);
        k_edge_expsum<<<blk((long long)E * HEADS, B), B, 0, stream>>>(dst, ews, mmax, ssum, E * HEADS);
        long long tot = (long long)E * HDl;
        k_edge_scatter<<<blk(tot, B), B, 0, stream>>>(src, dst, ews, ssum, gfc, gStride, D, accbuf, tot);
    };

    // ---- layer 1: fc(WMMA) -> attention -> scatter -> +b1, ELU ----
    k_gemm_bf16<4><<<gemm_blocks(HD / 64), 128, 0, stream>>>(xb, w1t, g, M16, KP1, HD);
    edge_phase(g, HD, HID, al1, ar1, acc);
    k_epilogue<<<blk((long long)N * HD, B), B, 0, stream>>>(h1, acc, (const float*)nullptr, 0, b1, HD, 1, (long long)N * HD);

    // ---- layer 2: identity residual ----
    k_cvt_pad<<<blk((long long)N * HD, B), B, 0, stream>>>(h1, hb, HD, HD, (long long)N * HD);
    k_gemm_bf16<4><<<gemm_blocks(HD / 64), 128, 0, stream>>>(hb, w2t, g, M16, HD, HD);
    edge_phase(g, HD, HID, al2, ar2, acc);
    k_epilogue<<<blk((long long)N * HD, B), B, 0, stream>>>(h2, acc, h1, HD, b2, HD, 1, (long long)N * HD);

    // ---- layer 3: fused [W3|resW3] GEMM (Nout=32), linear residual, no act ----
    k_cvt_pad<<<blk((long long)N * HD, B), B, 0, stream>>>(h2, hb, HD, HD, (long long)N * HD);
    k_gemm_bf16<2><<<gemm_blocks((2 * HO) / 32), 128, 0, stream>>>(hb, w3t, g, M16, HD, 2 * HO);
    edge_phase(g, 2 * HO, OUT, al3, ar3, acc);
    k_epilogue<<<blk((long long)N * HO, B), B, 0, stream>>>(h3, acc, g + HO, 2 * HO, b3, HO, 0, (long long)N * HO);

    // ---- mean over heads -> d_out [N,2] ----
    k_mean<<<blk((long long)N * OUT, B), B, 0, stream>>>(h3, (float*)d_out, OUT, N * OUT);
}
